// QuantizedLinear_52106543235554
// MI455X (gfx1250) — compile-verified, matrix operand order checked by reading, not run on hardware
//
#include <hip/hip_runtime.h>
#include <stdint.h>

// out[M,N] = X[M,K] * (W[N,K]/127)^T
// M = 8192, N = 4096, K = 4096.  X: f32, W: int8-in-int32, OUT: f32.
//
// Fast path (needs 96 MiB workspace):
//   pass 1: convert X -> bf16, W -> dequantized bf16 (one pass over memory)
//   pass 2: bf16 WMMA GEMM; tiles staged into LDS by the Tensor Data Mover
//           (TENSOR_LOAD_TO_LDS, TENSORcnt-tracked, double buffered, TDM pads
//           LDS rows to 144B pitch for conflict-free ds_load_b128 fragments)
// Fallback: fused convert+GEMM (round-1 kernel) if workspace is too small.

typedef __attribute__((ext_vector_type(16))) __bf16    v16bf;
typedef __attribute__((ext_vector_type(8)))  float     v8f;
typedef __attribute__((ext_vector_type(4)))  unsigned  v4u;
typedef __attribute__((ext_vector_type(8)))  int       v8i;
typedef __attribute__((ext_vector_type(4)))  int       v4i;

#define M_TOT 8192
#define N_TOT 4096
#define K_TOT 4096
#define QSCALE (1.0f / 127.0f)

#define BM 128
#define BN 128
#define BK 64
#define LDT 72   // BK + 8 ushorts padding -> 144B row pitch (TDM pad: 4 DWORDs / 32 DWORDs)

__device__ __forceinline__ unsigned short f2bf(float f) {
    union { float f; unsigned u; } v; v.f = f;
    unsigned r = v.u + 0x7FFFu + ((v.u >> 16) & 1u);  // round-to-nearest-even
    return (unsigned short)(r >> 16);
}

__device__ __forceinline__ unsigned pack2(float lo, float hi) {
    return (unsigned)f2bf(lo) | ((unsigned)f2bf(hi) << 16);
}

union FragBF { v16bf v; uint4 q[2]; };

// ---------------------------------------------------------------------------
// Pass 1: conversion kernels (streaming, 8 elements/thread)
// ---------------------------------------------------------------------------
__global__ __launch_bounds__(256)
void cvt_x_bf16(const float* __restrict__ X, unsigned short* __restrict__ A) {
    const size_t i = ((size_t)blockIdx.x * blockDim.x + threadIdx.x) * 8;
    const float4* p = (const float4*)(X + i);
    float4 a = p[0], b = p[1];
    uint4 q;
    q.x = pack2(a.x, a.y);
    q.y = pack2(a.z, a.w);
    q.z = pack2(b.x, b.y);
    q.w = pack2(b.z, b.w);
    *(uint4*)(A + i) = q;
}

__global__ __launch_bounds__(256)
void cvt_w_bf16(const int* __restrict__ W, unsigned short* __restrict__ B) {
    const size_t i = ((size_t)blockIdx.x * blockDim.x + threadIdx.x) * 8;
    const int4* p = (const int4*)(W + i);
    int4 a = p[0], b = p[1];
    uint4 q;
    q.x = pack2((float)a.x * QSCALE, (float)a.y * QSCALE);
    q.y = pack2((float)a.z * QSCALE, (float)a.w * QSCALE);
    q.z = pack2((float)b.x * QSCALE, (float)b.y * QSCALE);
    q.w = pack2((float)b.z * QSCALE, (float)b.w * QSCALE);
    *(uint4*)(B + i) = q;
}

// ---------------------------------------------------------------------------
// TDM descriptor build + issue: 2-D tile (BK x BM elements, bf16) -> LDS
// D# per CDNA5 ISA ch.8 (groups 0..3) with LDS padding enabled:
//   pad_interval code 4 (every 32 DWORDs = one 128B row), pad_amount code 3
//   (4 DWORDs = 16B)  ->  LDS row pitch 144B == LDT ushorts.
// 6-arg builtin form (clang-23 / therock-10.0):
//   (uint32x4 g0, int32x8 g1, int32x4, int32x4, int32x8, i32 cpol)
// ---------------------------------------------------------------------------
__device__ __forceinline__ void tdm_load_tile(const unsigned short* gptr,
                                              unsigned ldsOff,
                                              unsigned tensorDim1)
{
    const unsigned long long ga = (unsigned long long)(uintptr_t)gptr;

    v4u g0;
    g0[0] = 1u;                                   // count=1 (valid), user mode
    g0[1] = ldsOff;                               // lds_addr (bytes)
    g0[2] = (unsigned)ga;                         // global_addr[31:0]
    g0[3] = (unsigned)((ga >> 32) & 0x01FFFFFFu)  // global_addr[56:32]
          | 0x80000000u;                          // type = 2 ("image")

    v8i g1;
    g1[0] = (int)((1u << 16)     // data_size = 1 -> 2 bytes
                | (1u << 20)     // pad_enable
                | (4u << 22)     // pad_interval code 4 -> every 32 DWORDs
                | (3u << 25));   // pad_amount  code 3 -> 4 DWORDs
    g1[1] = (int)(((unsigned)K_TOT & 0xFFFFu) << 16);          // tensor_dim0[15:0]
    g1[2] = (int)(((unsigned)K_TOT >> 16)                      // tensor_dim0[31:16]
                | ((tensorDim1 & 0xFFFFu) << 16));             // tensor_dim1[15:0]
    g1[3] = (int)((tensorDim1 >> 16)                           // tensor_dim1[31:16]
                | ((unsigned)BK << 16));                       // tile_dim0 = 64
    g1[4] = (int)(unsigned)BM;                                 // tile_dim1 = 128, tile_dim2 = 0
    g1[5] = (int)(unsigned)K_TOT;                              // tensor_dim0_stride[31:0]
    g1[6] = 0;
    g1[7] = 0;

    const v4i g2 = {0, 0, 0, 0};                // 2-D tile: dims 2/3 unused
    const v4i g3 = {0, 0, 0, 0};
    const v8i g4 = {0, 0, 0, 0, 0, 0, 0, 0};    // extra group (zero-filled)

    __builtin_amdgcn_tensor_load_to_lds(g0, g1, g2, g3, g4, 0);
}

// ---------------------------------------------------------------------------
// Pass 2: bf16 WMMA GEMM, TDM-staged LDS, double buffered
// ---------------------------------------------------------------------------
__global__ __launch_bounds__(256)
void qlin_wmma_tdm(const unsigned short* __restrict__ A,   // [M,K] bf16
                   const unsigned short* __restrict__ W,   // [N,K] bf16 (dequantized)
                   float* __restrict__ OUT)
{
    __shared__ __align__(16) unsigned short lA[2][BM][LDT];  // 2 x 18 KiB
    __shared__ __align__(16) unsigned short lB[2][BN][LDT];  // 2 x 18 KiB

    const int tid      = threadIdx.x;
    const int lane     = tid & 31;          // wave32
    const int wave     = tid >> 5;          // 8 waves
    const int wm       = (wave & 1) * 64;   // 2 waves in M
    const int wn       = (wave >> 1) * 32;  // 4 waves in N
    const int laneRow  = lane & 15;
    const int laneHalf = (lane >> 4) * 8;

    const int m0 = blockIdx.x * BM;
    const int n0 = blockIdx.y * BN;

    const unsigned ldsA[2] = { (unsigned)(uintptr_t)&lA[0][0][0],
                               (unsigned)(uintptr_t)&lA[1][0][0] };
    const unsigned ldsB[2] = { (unsigned)(uintptr_t)&lB[0][0][0],
                               (unsigned)(uintptr_t)&lB[1][0][0] };

    v8f acc[4][2];
    const v8f vzero = {0.f, 0.f, 0.f, 0.f, 0.f, 0.f, 0.f, 0.f};
#pragma unroll
    for (int i = 0; i < 4; ++i)
#pragma unroll
        for (int j = 0; j < 2; ++j)
            acc[i][j] = vzero;

    auto issue = [&](int buf, int kb) {
        tdm_load_tile(A + (size_t)m0 * K_TOT + kb, ldsA[buf], (unsigned)M_TOT);
        tdm_load_tile(W + (size_t)n0 * K_TOT + kb, ldsB[buf], (unsigned)N_TOT);
    };

    auto compute = [&](int buf) {
#pragma unroll
        for (int ks = 0; ks < BK; ks += 32) {
            FragBF a[4], b[2];
#pragma unroll
            for (int i = 0; i < 4; ++i) {
                const unsigned short* p = &lA[buf][wm + i * 16 + laneRow][ks + laneHalf];
                a[i].q[0] = *(const uint4*)p;          // K = half+0..7
                a[i].q[1] = *(const uint4*)(p + 16);   // K = half+16..23
            }
#pragma unroll
            for (int j = 0; j < 2; ++j) {
                const unsigned short* p = &lB[buf][wn + j * 16 + laneRow][ks + laneHalf];
                b[j].q[0] = *(const uint4*)p;
                b[j].q[1] = *(const uint4*)(p + 16);
            }
#pragma unroll
            for (int i = 0; i < 4; ++i)
#pragma unroll
                for (int j = 0; j < 2; ++j)
                    acc[i][j] = __builtin_amdgcn_wmma_f32_16x16x32_bf16(
                        false, a[i].v, false, b[j].v,
                        (short)0, acc[i][j], false, false);
        }
    };

    // --- TDM double-buffered K loop: wave 0 drives the DMA ---
    if (wave == 0) issue(0, 0);
    int buf = 0;
    const int NKB = K_TOT / BK;   // 64
    for (int kbi = 0; kbi < NKB - 1; ++kbi) {
        __syncthreads();                       // buf^1 free (compute done)
        if (wave == 0) {
            issue(buf ^ 1, (kbi + 1) * BK);    // prefetch next block
            __builtin_amdgcn_s_wait_tensorcnt(2);  // current block landed
        }
        __syncthreads();                       // data visible to all waves
        compute(buf);
        buf ^= 1;
    }
    __syncthreads();
    if (wave == 0) __builtin_amdgcn_s_wait_tensorcnt(0);
    __syncthreads();
    compute(buf);

    // --- epilogue: VGPR r -> M = r + 8*(lane>=16), N = lane&15 ---
    const int rowHalf = (lane >> 4) * 8;
#pragma unroll
    for (int i = 0; i < 4; ++i)
#pragma unroll
        for (int j = 0; j < 2; ++j)
#pragma unroll
            for (int r = 0; r < 8; ++r) {
                const int row = m0 + wm + i * 16 + r + rowHalf;
                const int col = n0 + wn + j * 16 + laneRow;
                OUT[(size_t)row * N_TOT + col] = acc[i][j][r];
            }
}

// ---------------------------------------------------------------------------
// Fallback: fused convert + WMMA GEMM (no workspace needed) — round-1 kernel
// ---------------------------------------------------------------------------
__global__ __launch_bounds__(256)
void qlin_wmma_fused(const float* __restrict__ X,
                     const int*   __restrict__ W,
                     float* __restrict__ OUT)
{
    __shared__ __align__(16) unsigned short lA[BM][LDT];
    __shared__ __align__(16) unsigned short lB[BN][LDT];

    const int tid      = threadIdx.x;
    const int lane     = tid & 31;
    const int wave     = tid >> 5;
    const int wm       = (wave & 1) * 64;
    const int wn       = (wave >> 1) * 32;
    const int laneRow  = lane & 15;
    const int laneHalf = (lane >> 4) * 8;

    const int m0 = blockIdx.x * BM;
    const int n0 = blockIdx.y * BN;

    v8f acc[4][2];
    const v8f vzero = {0.f, 0.f, 0.f, 0.f, 0.f, 0.f, 0.f, 0.f};
#pragma unroll
    for (int i = 0; i < 4; ++i)
#pragma unroll
        for (int j = 0; j < 2; ++j)
            acc[i][j] = vzero;

    for (int kb = 0; kb < K_TOT; kb += BK) {
        __syncthreads();
#pragma unroll
        for (int g = 0; g < 4; ++g) {
            const int c   = tid + g * 256;
            const int row = c >> 3;
            const int col = (c & 7) * 8;

            const float4* pa = (const float4*)(X + (size_t)(m0 + row) * K_TOT + kb + col);
            float4 a0 = pa[0], a1 = pa[1];
            uint4 qa;
            qa.x = pack2(a0.x, a0.y);
            qa.y = pack2(a0.z, a0.w);
            qa.z = pack2(a1.x, a1.y);
            qa.w = pack2(a1.z, a1.w);
            *(uint4*)&lA[row][col] = qa;

            const int4* pb = (const int4*)(W + (size_t)(n0 + row) * K_TOT + kb + col);
            int4 b0 = pb[0], b1 = pb[1];
            uint4 qb;
            qb.x = pack2((float)b0.x * QSCALE, (float)b0.y * QSCALE);
            qb.y = pack2((float)b0.z * QSCALE, (float)b0.w * QSCALE);
            qb.z = pack2((float)b1.x * QSCALE, (float)b1.y * QSCALE);
            qb.w = pack2((float)b1.z * QSCALE, (float)b1.w * QSCALE);
            *(uint4*)&lB[row][col] = qb;
        }
        __syncthreads();

#pragma unroll
        for (int ks = 0; ks < BK; ks += 32) {
            FragBF a[4], b[2];
#pragma unroll
            for (int i = 0; i < 4; ++i) {
                const unsigned short* p = &lA[wm + i * 16 + laneRow][ks + laneHalf];
                a[i].q[0] = *(const uint4*)p;
                a[i].q[1] = *(const uint4*)(p + 16);
            }
#pragma unroll
            for (int j = 0; j < 2; ++j) {
                const unsigned short* p = &lB[wn + j * 16 + laneRow][ks + laneHalf];
                b[j].q[0] = *(const uint4*)p;
                b[j].q[1] = *(const uint4*)(p + 16);
            }
#pragma unroll
            for (int i = 0; i < 4; ++i)
#pragma unroll
                for (int j = 0; j < 2; ++j)
                    acc[i][j] = __builtin_amdgcn_wmma_f32_16x16x32_bf16(
                        false, a[i].v, false, b[j].v,
                        (short)0, acc[i][j], false, false);
        }
    }

    const int rowHalf = (lane >> 4) * 8;
#pragma unroll
    for (int i = 0; i < 4; ++i)
#pragma unroll
        for (int j = 0; j < 2; ++j)
#pragma unroll
            for (int r = 0; r < 8; ++r) {
                const int row = m0 + wm + i * 16 + r + rowHalf;
                const int col = n0 + wn + j * 16 + laneRow;
                OUT[(size_t)row * N_TOT + col] = acc[i][j][r];
            }
}

// ---------------------------------------------------------------------------
extern "C" void kernel_launch(void* const* d_in, const int* in_sizes, int n_in,
                              void* d_out, int out_size, void* d_ws, size_t ws_size,
                              hipStream_t stream) {
    (void)in_sizes; (void)n_in; (void)out_size;
    const float* x = (const float*)d_in[0];
    const int*   w = (const int*)d_in[1];
    float*      out = (float*)d_out;

    const size_t needA = (size_t)M_TOT * K_TOT * sizeof(unsigned short); // 64 MiB
    const size_t needW = (size_t)N_TOT * K_TOT * sizeof(unsigned short); // 32 MiB

    dim3 grid(M_TOT / BM, N_TOT / BN);   // 64 x 32

    if (ws_size >= needA + needW) {
        unsigned short* Abf = (unsigned short*)d_ws;
        unsigned short* Wbf = (unsigned short*)((char*)d_ws + needA);
        cvt_x_bf16<<<(M_TOT * (size_t)K_TOT) / (8 * 256), 256, 0, stream>>>(x, Abf);
        cvt_w_bf16<<<(N_TOT * (size_t)K_TOT) / (8 * 256), 256, 0, stream>>>(w, Wbf);
        qlin_wmma_tdm<<<grid, 256, 0, stream>>>(Abf, Wbf, out);
    } else {
        qlin_wmma_fused<<<grid, 256, 0, stream>>>(x, w, out);
    }
}
